// MinimalMACE_81088982548516
// MI455X (gfx1250) — matrix operand order repaired and checked by reading, not compiled, not run on hardware
//
#include <hip/hip_runtime.h>
#include <hip/hip_bf16.h>
#include <math.h>

typedef __attribute__((ext_vector_type(16))) _Float16 v16h;
typedef __attribute__((ext_vector_type(8)))  float    v8f;

// ---------------------------------------------------------------------------
// WMMA helpers (CDNA5 gfx1250, wave32)
// ---------------------------------------------------------------------------
static __device__ __forceinline__ v8f wmma16(v16h a, v16h b, v8f c) {
    // D = A(16x32 f16) * B(32x16 f16) + C(16x16 f32)
    return __builtin_amdgcn_wmma_f32_16x16x32_f16(
        /*neg_a=*/false, a, /*neg_b=*/false, b,
        /*c_mod=*/(short)0, c, /*reuse_a=*/false, /*reuse_b=*/false);
}

// A-fragment gather from LDS activations stored row-major [16][rowStride] halfs.
// Contiguous k order used by this file (packer matches): lane L: M=L%16, g=L/16; elem e -> K = g*16 + e
static __device__ __forceinline__ v16h lds_afrag(const _Float16* base, int rowStride, int kbase) {
    int lane = threadIdx.x & 31;
    int M = lane & 15;
    int g = lane >> 4;
    v16h a;
#pragma unroll
    for (int e = 0; e < 16; ++e) {
        int K = kbase + (g << 4) + e;
        a[e] = base[M * rowStride + K];
    }
    return a;
}

// B-fragment: weights pre-packed so each lane's 16 halfs are contiguous (32B aligned).
static __device__ __forceinline__ v16h gl_bfrag(const _Float16* packed, int tile) {
    int lane = threadIdx.x & 31;
    return *(const v16h*)(packed + (size_t)tile * 512 + lane * 16);
}

static __device__ __forceinline__ float silu_f(float v) {
    return v / (1.0f + __expf(-v));
}

static __device__ __forceinline__ int l_of_lm(int lm) {
    return (lm == 0) ? 0 : (lm < 4) ? 1 : ((lm < 9) ? 2 : 3);
}

// Async global->LDS 16-byte copy (ASYNCcnt path, ISA 10/15.18).
// ldsaddr = wave-relative LDS byte address (low 32 bits of generic pointer).
static __device__ __forceinline__ void async_copy_b128(void* lds_ptr, const void* gptr) {
    unsigned lds = (unsigned)(size_t)lds_ptr;
    unsigned long long ga = (unsigned long long)(size_t)gptr;
    asm volatile("global_load_async_to_lds_b128 %0, %1, off"
                 :: "v"(lds), "v"(ga) : "memory");
}
static __device__ __forceinline__ void async_wait0() {
    asm volatile("s_wait_asynccnt 0x0" ::: "memory");
}

// ---------------------------------------------------------------------------
// Weight packer: src f32 row-major [Ksrc][Nsrc] -> f16 B-fragments.
// dst[tile*512 + lane*16 + e] = src[kt*32 + (lane/16)*16 + e][nt*16 + lane%16]
// grid = Kt*Nt blocks, 512 threads.
// ---------------------------------------------------------------------------
__global__ void k_pack(const float* __restrict__ src, _Float16* __restrict__ dst,
                       int Ksrc, int Nsrc, int Nt) {
    int tile = blockIdx.x;
    int kt = tile / Nt, nt = tile % Nt;
    int lane = threadIdx.x >> 4;   // 0..31
    int e = threadIdx.x & 15;
    int K = kt * 32 + ((lane >> 4) << 4) + e;
    int Ncol = nt * 16 + (lane & 15);
    float v = (K < Ksrc && Ncol < Nsrc) ? src[K * Nsrc + Ncol] : 0.0f;
    dst[(size_t)tile * 512 + lane * 16 + e] = (_Float16)v;
}

// ---------------------------------------------------------------------------
// Edge geometry: spherical harmonics Y[E][16] and radial basis EB[E][8]
// ---------------------------------------------------------------------------
__global__ void k_geom(const float* __restrict__ pos, const int* __restrict__ ei,
                       float* __restrict__ Y, float* __restrict__ EB, int E) {
    int e = blockIdx.x * blockDim.x + threadIdx.x;
    if (e >= E) return;
    int s = ei[e], d = ei[E + e];
    float vx = pos[d * 3 + 0] - pos[s * 3 + 0];
    float vy = pos[d * 3 + 1] - pos[s * 3 + 1];
    float vz = pos[d * 3 + 2] - pos[s * 3 + 2];
    float r = sqrtf(vx * vx + vy * vy + vz * vz + 1e-12f);
    float ir = 1.0f / r;
    float x = vx * ir, y = vy * ir, z = vz * ir;
    float x2 = x * x, y2 = y * y, z2 = z * z;
    const float s3 = 1.73205080757f, s15 = 3.87298334621f, s5 = 2.2360679775f;
    const float s105 = 10.2469507659f, s35_8 = 2.09165006634f;
    const float s21_8 = 1.62018517461f, s7 = 2.64575131106f;
    float* Ye = Y + (size_t)e * 16;
    Ye[0] = 1.0f;
    Ye[1] = s3 * x;  Ye[2] = s3 * y;  Ye[3] = s3 * z;
    Ye[4] = s15 * x * y;  Ye[5] = s15 * y * z;
    Ye[6] = 0.5f * s5 * (3.f * z2 - 1.f);
    Ye[7] = s15 * x * z;  Ye[8] = 0.5f * s15 * (x2 - y2);
    Ye[9]  = s35_8 * y * (3.f * x2 - y2);
    Ye[10] = s105 * x * y * z;
    Ye[11] = s21_8 * y * (5.f * z2 - 1.f);
    Ye[12] = 0.5f * s7 * z * (5.f * z2 - 3.f);
    Ye[13] = s21_8 * x * (5.f * z2 - 1.f);
    Ye[14] = 0.5f * s105 * z * (x2 - y2);
    Ye[15] = s35_8 * x * (3.f * x2 - y2);
    // radial: bessel * polynomial cutoff (R_MAX=5, P_CUT=5)
    float xx = r * 0.2f;
    float x5 = xx * xx * xx * xx * xx;
    float fcut = 1.f - 21.f * x5 + 35.f * x5 * xx - 15.f * x5 * xx * xx;
    if (xx >= 1.f) fcut = 0.f;
    const float amp = 0.632455532034f;       // sqrt(2/5)
    const float w0 = 0.62831853072f;         // pi/5
    float* Eb = EB + (size_t)e * 8;
#pragma unroll
    for (int n = 1; n <= 8; ++n)
        Eb[n - 1] = amp * __sinf((float)n * w0 * r) * ir * fcut;
}

// ---------------------------------------------------------------------------
// Embedding: h = x @ W_embed; also recover element index (x is one-hot)
// ---------------------------------------------------------------------------
__global__ void k_embed(const float* __restrict__ x, const float* __restrict__ Wemb,
                        float* __restrict__ H, int* __restrict__ ELEM, int N) {
    int n = blockIdx.x;
    int k = threadIdx.x;
    float acc = 0.f;
#pragma unroll
    for (int e = 0; e < 10; ++e) acc += x[n * 10 + e] * Wemb[e * 64 + k];
    H[(size_t)n * 64 + k] = acc;
    if (k == 0) {
        int best = 0; float bv = x[n * 10];
#pragma unroll
        for (int e = 1; e < 10; ++e) { float v = x[n * 10 + e]; if (v > bv) { bv = v; best = e; } }
        ELEM[n] = best;
    }
}

__global__ void k_zero(float* p, size_t n) {
    size_t i = (size_t)blockIdx.x * blockDim.x + threadIdx.x;
    if (i < n) p[i] = 0.f;
}

// ---------------------------------------------------------------------------
// Node GEMM: OUT(N,64) = IN(N,64) @ Wp(packed 64x64), optional per-element skip
// 128 threads (4 waves) per 16-node tile; each wave owns one 16x16 j-tile.
// ---------------------------------------------------------------------------
__global__ void k_node_gemm(const float* __restrict__ IN, const _Float16* __restrict__ Wp,
                            float* __restrict__ OUT, int N,
                            const float* __restrict__ skipH, const float* __restrict__ Wsc,
                            const int* __restrict__ ELEM) {
    __shared__ __align__(16) _Float16 sA[16 * 64];
    __shared__ __align__(16) float    sH[16 * 64];
    int nt = blockIdx.x;
    int tid = threadIdx.x;
    // skip-h tile: verbatim f32 copy -> async global->LDS (ASYNCcnt path)
    if (skipH) {
#pragma unroll
        for (int i = 0; i < 2; ++i) {
            int idx = tid + i * 128;                 // 256 x 16B transfers
            int row = idx >> 4, c4 = (idx & 15) << 2;
            async_copy_b128(&sH[row * 64 + c4],
                            skipH + ((size_t)nt * 16 + row) * 64 + c4);
        }
    }
    // activation tile needs f32->f16 conversion: manual stage
#pragma unroll
    for (int i = 0; i < 8; ++i) {
        int idx = tid + i * 128;
        int row = idx >> 6, col = idx & 63;
        float v = IN[((size_t)nt * 16 + row) * 64 + col];
        sA[idx] = (_Float16)v;
    }
    if (skipH) async_wait0();
    __syncthreads();
    int w = tid >> 5;
    int lane = tid & 31, g = lane >> 4, cl = lane & 15;
    v8f acc = {};
#pragma unroll
    for (int kt = 0; kt < 2; ++kt)
        acc = wmma16(lds_afrag(sA, 64, kt * 32), gl_bfrag(Wp, kt * 4 + w), acc);
    int col = w * 16 + cl;
#pragma unroll
    for (int v = 0; v < 8; ++v) {
        int row = v + g * 8;
        int n = nt * 16 + row;
        float val = acc[v];
        if (skipH) {
            const float* wsc = Wsc + (size_t)ELEM[n] * 4096;
            float s = 0.f;
            for (int k = 0; k < 64; ++k) s += sH[row * 64 + k] * wsc[k * 64 + col];
            val += s;
        }
        OUT[(size_t)n * 64 + col] = val;
    }
}

// ---------------------------------------------------------------------------
// Fused edge MLP + message scatter (per layer).
// 128 threads (4 waves) per 16-edge tile.
//   t1=silu(eb@R1); t2=silu(t1@R2); t3=silu(t2@R3); tpw=t3@R4 (in LDS, f32)
//   msg[e,k,lm] = tpw[e, 4k+L(lm)] * hup[src,k] * Y[e,lm] / 16  -> atomicAdd agg[dst]
// ---------------------------------------------------------------------------
__global__ void k_edge(const float* __restrict__ EB, const float* __restrict__ Y,
                       const int* __restrict__ ei, const float* __restrict__ HUP,
                       float* __restrict__ AGG,
                       const _Float16* __restrict__ R1p, const _Float16* __restrict__ R2p,
                       const _Float16* __restrict__ R3p, const _Float16* __restrict__ R4p,
                       int E) {
    __shared__ __align__(16) _Float16 sIn[16 * 64];
    __shared__ __align__(16) _Float16 sT[16 * 64];
    __shared__ __align__(16) float    sTPW[16 * 256];
    __shared__ __align__(16) float    sY[16 * 16];
    __shared__ __align__(16) float    sHUP[16 * 64];
    __shared__ int      sSrc[16];
    __shared__ int      sDst[16];
    int et = blockIdx.x;
    int tid = threadIdx.x;
    int w = tid >> 5, lane = tid & 31, g = lane >> 4, cl = lane & 15;

    // prefetch next tile's radial/harmonic rows (global_prefetch_b8, speculative)
    __builtin_prefetch(EB + ((size_t)(et + 1) * 16) * 8, 0, 3);
    __builtin_prefetch(Y + ((size_t)(et + 1) * 16) * 16, 0, 3);

    // stage 0: EB tile -> sIn (16 x 32, K padded with zeros)
#pragma unroll
    for (int i = 0; i < 4; ++i) {
        int idx = tid + i * 128;
        int row = idx >> 5, col = idx & 31;
        float v = (col < 8) ? EB[((size_t)et * 16 + row) * 8 + col] : 0.f;
        sIn[row * 32 + col] = (_Float16)v;
    }
    if (tid < 16) {
        sSrc[tid] = ei[et * 16 + tid];
        sDst[tid] = ei[E + et * 16 + tid];
    }
    // Y tile: verbatim f32 copy -> async global->LDS (64 x 16B)
    if (tid < 64) {
        int row = tid >> 2, c4 = (tid & 3) << 2;
        async_copy_b128(&sY[row * 16 + c4],
                        Y + ((size_t)et * 16 + row) * 16 + c4);
    }
    __syncthreads();

    // gathered hup rows (needs sSrc): async global->LDS, 256 x 16B
#pragma unroll
    for (int i = 0; i < 2; ++i) {
        int idx = tid + i * 128;
        int row = idx >> 4, c4 = (idx & 15) << 2;
        async_copy_b128(&sHUP[row * 64 + c4],
                        HUP + (size_t)sSrc[row] * 64 + c4);
    }

    // stage 1: t1 = silu(eb @ R1)  (K=32 padded, 1 WMMA per wave)
    {
        v8f acc = {};
        acc = wmma16(lds_afrag(sIn, 32, 0), gl_bfrag(R1p, w), acc);
        int col = w * 16 + cl;
#pragma unroll
        for (int v = 0; v < 8; ++v)
            sT[(v + g * 8) * 64 + col] = (_Float16)silu_f(acc[v]);
    }
    __syncthreads();

    // stage 2: t2 = silu(t1 @ R2)
    {
        v8f acc = {};
#pragma unroll
        for (int kt = 0; kt < 2; ++kt)
            acc = wmma16(lds_afrag(sT, 64, kt * 32), gl_bfrag(R2p, kt * 4 + w), acc);
        int col = w * 16 + cl;
#pragma unroll
        for (int v = 0; v < 8; ++v)
            sIn[(v + g * 8) * 64 + col] = (_Float16)silu_f(acc[v]);
    }
    __syncthreads();

    // stage 3: t3 = silu(t2 @ R3)
    {
        v8f acc = {};
#pragma unroll
        for (int kt = 0; kt < 2; ++kt)
            acc = wmma16(lds_afrag(sIn, 64, kt * 32), gl_bfrag(R3p, kt * 4 + w), acc);
        int col = w * 16 + cl;
#pragma unroll
        for (int v = 0; v < 8; ++v)
            sT[(v + g * 8) * 64 + col] = (_Float16)silu_f(acc[v]);
    }
    __syncthreads();

    // stage 4: tpw = t3 @ R4 (64 -> 256); each wave does 4 column tiles
#pragma unroll
    for (int jj = 0; jj < 4; ++jj) {
        int jt = w * 4 + jj;
        v8f acc = {};
#pragma unroll
        for (int kt = 0; kt < 2; ++kt)
            acc = wmma16(lds_afrag(sT, 64, kt * 32), gl_bfrag(R4p, kt * 16 + jt), acc);
        int col = jt * 16 + cl;
#pragma unroll
        for (int v = 0; v < 8; ++v)
            sTPW[(v + g * 8) * 256 + col] = acc[v];
    }
    async_wait0();     // sY + sHUP async copies complete (per wave)
    __syncthreads();   // sTPW + async-staged LDS visible to all waves

    // messages: 16 edges x 64 k x 16 lm, scatter with float atomics into L2-resident agg
    const float inv_avg = 0.0625f;  // 1/AVG_NEIGH
    for (int i = 0; i < 128; ++i) {
        int idx = tid + i * 128;
        int row = idx >> 10;
        int rem = idx & 1023;       // k*16 + lm
        int k = rem >> 4, lm = rem & 15;
        float v = sTPW[row * 256 + k * 4 + l_of_lm(lm)] * sHUP[row * 64 + k] *
                  sY[row * 16 + lm] * inv_avg;
        atomicAdd(&AGG[(size_t)sDst[row] * 1024 + rem], v);
    }
}

// ---------------------------------------------------------------------------
// A[n,j,m] = sum_k agg[n,k,m] * W_post[L(m)][k,j]  -- 16 batched 64x64 GEMMs.
// agg read layout [n][k*16+m]; A written in-place (aliased) as [n][m*64+j].
// Safe: all global reads complete before the barrier that precedes any write,
// and blocks touch disjoint node ranges.
// ---------------------------------------------------------------------------
__global__ void k_nodeA(float* __restrict__ AGG, const _Float16* __restrict__ WPp, int N) {
    __shared__ __align__(16) _Float16 sAgg[16 * 16 * 64];   // [m][row][k]
    int nt = blockIdx.x;
    int tid = threadIdx.x;                    // 256 threads, 8 waves
#pragma unroll
    for (int i = 0; i < 64; ++i) {
        int idx = tid + i * 256;              // 16384 elements
        int row = idx >> 10;
        int rem = idx & 1023;
        int k = rem >> 4, m = rem & 15;
        sAgg[(m * 16 + row) * 64 + k] = (_Float16)AGG[((size_t)nt * 16 + row) * 1024 + rem];
    }
    __syncthreads();
    int w = tid >> 5, lane = tid & 31, g = lane >> 4, cl = lane & 15;
    for (int tt = 0; tt < 8; ++tt) {
        int tile = w * 8 + tt;
        int m = tile >> 2, jt = tile & 3;
        int l = l_of_lm(m);
        v8f acc = {};
#pragma unroll
        for (int kt = 0; kt < 2; ++kt)
            acc = wmma16(lds_afrag(sAgg + m * 16 * 64, 64, kt * 32),
                         gl_bfrag(WPp + (size_t)l * 4096, kt * 4 + jt), acc);
        int col = jt * 16 + cl;
#pragma unroll
        for (int v = 0; v < 8; ++v) {
            int row = v + g * 8;
            AGG[((size_t)nt * 16 + row) * 1024 + m * 64 + col] = acc[v];
        }
    }
}

// ---------------------------------------------------------------------------
// Per-node contraction: A0, sq = sum_m A^2 / sqrt(2l+1), c[9], b = sum_p c_p * wp_p
// ---------------------------------------------------------------------------
__global__ void k_contract(const float* __restrict__ A, const float* __restrict__ Wprod,
                           const int* __restrict__ ELEM, float* __restrict__ B, int N) {
    int n = blockIdx.x * 4 + (threadIdx.x >> 6);
    int k = threadIdx.x & 63;
    if (n >= N) return;
    const float* An = A + (size_t)n * 1024;   // [m*64 + j], thread uses j=k
    float a0 = An[k];
    float sq0 = 0.f, sq1 = 0.f, sq2 = 0.f, sq3 = 0.f;
#pragma unroll
    for (int m = 0; m < 16; ++m) {
        float v = An[m * 64 + k];
        float v2 = v * v;
        int l = l_of_lm(m);
        if (l == 0) sq0 += v2; else if (l == 1) sq1 += v2;
        else if (l == 2) sq2 += v2; else sq3 += v2;
    }
    sq1 *= 0.57735026919f;   // 1/sqrt(3)
    sq2 *= 0.44721359550f;   // 1/sqrt(5)
    sq3 *= 0.37796447301f;   // 1/sqrt(7)
    float c[9] = { a0, sq0, sq1, sq2, sq3, sq0 * a0, sq1 * a0, sq2 * a0, sq3 * a0 };
    const float* wp = Wprod + (size_t)ELEM[n] * 9 * 64;
    float acc = 0.f;
#pragma unroll
    for (int p = 0; p < 9; ++p) acc += c[p] * wp[p * 64 + k];
    B[(size_t)n * 64 + k] = acc;
}

// ---------------------------------------------------------------------------
// Readout: prop = h @ W_ro (N,10); output = [prop | h]
// ---------------------------------------------------------------------------
__global__ void k_readout(const float* __restrict__ H, const float* __restrict__ Wro,
                          float* __restrict__ out, int N) {
    int n = blockIdx.x;
    int j = threadIdx.x;
    if (j < 10) {
        float acc = 0.f;
        for (int k = 0; k < 64; ++k) acc += H[(size_t)n * 64 + k] * Wro[k * 10 + j];
        out[(size_t)n * 10 + j] = acc;
    }
    out[(size_t)N * 10 + (size_t)n * 64 + j] = H[(size_t)n * 64 + j];
}

// ---------------------------------------------------------------------------
extern "C" void kernel_launch(void* const* d_in, const int* in_sizes, int n_in,
                              void* d_out, int out_size, void* d_ws, size_t ws_size,
                              hipStream_t stream) {
    (void)n_in; (void)out_size; (void)ws_size;
    const float* pos   = (const float*)d_in[0];
    const float* x     = (const float*)d_in[1];
    const int*   ei    = (const int*)  d_in[2];
    const float* Wemb  = (const float*)d_in[3];
    const float* Wup   = (const float*)d_in[4];
    const float* R1    = (const float*)d_in[5];
    const float* R2    = (const float*)d_in[6];
    const float* R3    = (const float*)d_in[7];
    const float* R4    = (const float*)d_in[8];
    const float* Wpost = (const float*)d_in[9];
    const float* Wsc   = (const float*)d_in[10];
    const float* Wprod = (const float*)d_in[11];
    const float* Wout  = (const float*)d_in[12];
    const float* Wro   = (const float*)d_in[13];
    int N = in_sizes[0] / 3;
    int E = in_sizes[2] / 2;

    // workspace layout (all chunk sizes keep 32B alignment for packed f16 frags)
    float* ws = (float*)d_ws;
    size_t off = 0;
    float* Yb  = ws + off; off += (size_t)E * 16;
    float* EBb = ws + off; off += (size_t)E * 8;
    float* HUP = ws + off; off += (size_t)N * 64;
    float* H0  = ws + off; off += (size_t)N * 64;
    float* H1  = ws + off; off += (size_t)N * 64;
    float* BB  = ws + off; off += (size_t)N * 64;
    float* AGG = ws + off; off += (size_t)N * 1024;   // aliased as A after k_nodeA
    _Float16* PACK = (_Float16*)(ws + off); off += 51200;  // 2 layers x 51200 halfs
    int* ELEM = (int*)(ws + off); off += (size_t)N;

    // per-layer packed-weight offsets (halfs)
    const size_t LSTRIDE = 51200;
    const size_t R1P = 0, R2P = 2048, R3P = 6144, R4P = 10240;
    const size_t WUPP = 26624, WPOSTP = 30720, WOUTP = 47104;

    for (int i = 0; i < 2; ++i) {
        _Float16* LP = PACK + (size_t)i * LSTRIDE;
        k_pack<<<4,  512, 0, stream>>>(R1 + (size_t)i * 8 * 64,   LP + R1P,  8,  64, 4);
        k_pack<<<8,  512, 0, stream>>>(R2 + (size_t)i * 64 * 64,  LP + R2P, 64,  64, 4);
        k_pack<<<8,  512, 0, stream>>>(R3 + (size_t)i * 64 * 64,  LP + R3P, 64,  64, 4);
        k_pack<<<32, 512, 0, stream>>>(R4 + (size_t)i * 64 * 256, LP + R4P, 64, 256, 16);
        k_pack<<<8,  512, 0, stream>>>(Wup + (size_t)i * 64 * 64, LP + WUPP, 64, 64, 4);
        for (int l = 0; l < 4; ++l)
            k_pack<<<8, 512, 0, stream>>>(Wpost + ((size_t)i * 4 + l) * 64 * 64,
                                          LP + WPOSTP + (size_t)l * 4096, 64, 64, 4);
        k_pack<<<8,  512, 0, stream>>>(Wout + (size_t)i * 64 * 64, LP + WOUTP, 64, 64, 4);
    }

    k_geom <<<(E + 255) / 256, 256, 0, stream>>>(pos, ei, Yb, EBb, E);
    k_embed<<<N, 64, 0, stream>>>(x, Wemb, H0, ELEM, N);

    float* hcur = H0;
    float* hnxt = H1;
    for (int i = 0; i < 2; ++i) {
        _Float16* LP = PACK + (size_t)i * LSTRIDE;
        k_node_gemm<<<N / 16, 128, 0, stream>>>(hcur, LP + WUPP, HUP, N,
                                                nullptr, nullptr, ELEM);
        size_t aggN = (size_t)N * 1024;
        k_zero<<<(int)((aggN + 255) / 256), 256, 0, stream>>>(AGG, aggN);
        k_edge<<<E / 16, 128, 0, stream>>>(EBb, Yb, ei, HUP, AGG,
                                           LP + R1P, LP + R2P, LP + R3P, LP + R4P, E);
        k_nodeA<<<N / 16, 256, 0, stream>>>(AGG, LP + WPOSTP, N);
        k_contract<<<(N + 3) / 4, 256, 0, stream>>>(AGG, Wprod + (size_t)i * 10 * 9 * 64,
                                                    ELEM, BB, N);
        k_node_gemm<<<N / 16, 128, 0, stream>>>(BB, LP + WOUTP, hnxt, N,
                                                (i > 0) ? hcur : nullptr,
                                                Wsc + (size_t)i * 10 * 64 * 64, ELEM);
        float* t = hcur; hcur = hnxt; hnxt = t;
    }

    k_readout<<<N, 64, 0, stream>>>(hcur, Wro, (float*)d_out, N);
}
